// QuantizedLinear_76175539962496
// MI455X (gfx1250) — matrix operand order repaired and checked by reading, not compiled, not run on hardware
//
#include <hip/hip_runtime.h>

// ---------------------------------------------------------------------------
// int4-blockwise-dequant linear  y = x @ W^T + b  for gfx1250 (MI455X).
//
// Roofline: 2*8192*11008*4096 = 0.74 TFLOP over ~0.68 GB unique traffic
// (~1090 FLOP/B) -> matrix-unit bound.  Round-1 lesson: fusing dequant into
// the GEMM re-converts every x element 86x and every weight 64x, giving
// ~128 VALU instr per 16 WMMAs per wave -> VALU-bound.  So:
//   pass 1: x f32->f16 once              (64 MB ws,  ~3 us @ 23.3 TB/s)
//   pass 2: W int4->f16 dequant once     (86 MB ws, ~12 us)
//   pass 3: f16 WMMA GEMM, LDS tiles filled by the Tensor Data Mover
//           (tensor_load_to_lds, TENSORcnt double-buffering, HW row padding)
// W-f16 (86 MB) fits in the 192 MB L2; grid order keeps an N-slab resident.
// This toolchain exposes the 6-arg tensor_load_to_lds builtin (clang-23).
// ---------------------------------------------------------------------------

#define IN_FEATURES  4096
#define OUT_FEATURES 11008
#define QBLOCK       64
#define M_TOTAL      8192         // 4 * 2048

#define BM 128
#define BN 128
#define BK 64
#define LDS_STRIDE 72             // halves: 144 B rows = 128 B data + 16 B pad

typedef __attribute__((ext_vector_type(16))) _Float16 v16h;
typedef __attribute__((ext_vector_type(8)))  _Float16 v8h;
typedef __attribute__((ext_vector_type(4)))  _Float16 v4h;
typedef __attribute__((ext_vector_type(8)))  float    v8f;
typedef unsigned int u32x4 __attribute__((ext_vector_type(4)));
typedef int          i32x8 __attribute__((ext_vector_type(8)));
typedef int          i32x4 __attribute__((ext_vector_type(4)));

#if __has_builtin(__builtin_amdgcn_tensor_load_to_lds)
#define USE_TDM 1
#else
#define USE_TDM 0
#endif

// ---------------------------------------------------------------- pass 1 ----
__global__ __launch_bounds__(256)
void cvt_x_kernel(const float* __restrict__ x, _Float16* __restrict__ xh)
{
    const size_t i = ((size_t)blockIdx.x * 256 + threadIdx.x) * 8;
    const float4 a = *(const float4*)(x + i);
    const float4 b = *(const float4*)(x + i + 4);
    v8h h;
    h[0] = (_Float16)a.x; h[1] = (_Float16)a.y; h[2] = (_Float16)a.z; h[3] = (_Float16)a.w;
    h[4] = (_Float16)b.x; h[5] = (_Float16)b.y; h[6] = (_Float16)b.z; h[7] = (_Float16)b.w;
    *(v8h*)(xh + i) = h;
}

// ---------------------------------------------------------------- pass 2 ----
__global__ __launch_bounds__(256)
void dequant_w_kernel(const int* __restrict__ wq, const float* __restrict__ absmax,
                      _Float16* __restrict__ wh)
{
    const size_t e = ((size_t)blockIdx.x * 256 + threadIdx.x) * 16;  // 16 codes/thread
    const float s = absmax[e >> 6] * (1.0f / 7.0f);                  // one qblock scale
    const int4 q0 = *(const int4*)(wq + e);
    const int4 q1 = *(const int4*)(wq + e + 4);
    const int4 q2 = *(const int4*)(wq + e + 8);
    const int4 q3 = *(const int4*)(wq + e + 12);
    v8h h0, h1;
    h0[0] = (_Float16)((float)q0.x * s); h0[1] = (_Float16)((float)q0.y * s);
    h0[2] = (_Float16)((float)q0.z * s); h0[3] = (_Float16)((float)q0.w * s);
    h0[4] = (_Float16)((float)q1.x * s); h0[5] = (_Float16)((float)q1.y * s);
    h0[6] = (_Float16)((float)q1.z * s); h0[7] = (_Float16)((float)q1.w * s);
    h1[0] = (_Float16)((float)q2.x * s); h1[1] = (_Float16)((float)q2.y * s);
    h1[2] = (_Float16)((float)q2.z * s); h1[3] = (_Float16)((float)q2.w * s);
    h1[4] = (_Float16)((float)q3.x * s); h1[5] = (_Float16)((float)q3.y * s);
    h1[6] = (_Float16)((float)q3.z * s); h1[7] = (_Float16)((float)q3.w * s);
    *(v8h*)(wh + e)     = h0;
    *(v8h*)(wh + e + 8) = h1;
}

// ----------------------------------------------------------- TDM helper ----
#if USE_TDM
// 2D tile load Global->LDS via the Tensor Data Mover (CDNA5 ISA).
// Rows of tile_d0 f16 elements; HW pads each 32-DWORD (128 B) row with
// 4 DWORDs (16 B) -> LDS row stride 144 B = LDS_STRIDE halves.
__device__ __forceinline__ void tdm_load_tile_f16(unsigned lds_off,
                                                  const _Float16* gptr,
                                                  unsigned tensor_d0,
                                                  unsigned tensor_d1,
                                                  unsigned tile_d0,
                                                  unsigned tile_d1,
                                                  unsigned stride0)
{
    const unsigned long long ga = (unsigned long long)(uintptr_t)gptr;
    u32x4 g0;
    g0[0] = 1u;                                            // count=1, user D#
    g0[1] = lds_off;                                       // LDS byte address
    g0[2] = (unsigned)(ga & 0xFFFFFFFFu);                  // global_addr lo
    g0[3] = (unsigned)((ga >> 32) & 0x01FFFFFFu) | (2u << 30);  // addr hi | type=2
    i32x8 g1;
    g1[0] = (int)((1u << 16)      // data_size = 2 B
                | (1u << 20)      // pad_enable
                | (4u << 22)      // pad_interval: 32 DWORDs between pads
                | (3u << 25));    // pad_amount: 4 DWORDs (16 B)
    g1[1] = (int)(tensor_d0 << 16);                        // tensor_dim0[15:0]
    g1[2] = (int)((tensor_d0 >> 16) | (tensor_d1 << 16));  // dim0 hi | dim1 lo
    g1[3] = (int)((tensor_d1 >> 16) | (tile_d0 << 16));    // dim1 hi | tile_dim0
    g1[4] = (int)(tile_d1 & 0xFFFFu);                      // tile_dim1, tile_dim2=0
    g1[5] = (int)stride0;                                  // tensor_dim0_stride lo
    g1[6] = 0;
    g1[7] = 0;
    const i32x4 z4 = {0, 0, 0, 0};                         // 2D: groups 2/3 unused
    const i32x8 z8 = {0, 0, 0, 0, 0, 0, 0, 0};
    __builtin_amdgcn_tensor_load_to_lds(g0, g1, z4, z4, z8, 0);
}
#endif

// ---------------------------------------------------------------- pass 3 ----
__global__ __launch_bounds__(256)
void qgemm_f16_wmma(const _Float16* __restrict__ xh,
                    const _Float16* __restrict__ wh,
                    const float*    __restrict__ bias,
                    float*          __restrict__ out)
{
    __shared__ _Float16 sA[2][BM * LDS_STRIDE];
    __shared__ _Float16 sB[2][BN * LDS_STRIDE];

    const int tid   = threadIdx.x;
    const int lane  = tid & 31;               // wave32
    const int wave  = tid >> 5;               // 8 waves
    const int waveM = wave >> 2;              // 0..1 -> 64 M rows
    const int waveN = wave & 3;               // 0..3 -> 32 N cols

    const int mTiles = M_TOTAL / BM;          // 64; M fastest -> W slab hot in L2
    const int nTile  = blockIdx.x / mTiles;
    const int mTile  = blockIdx.x % mTiles;
    const int m0 = mTile * BM;
    const int n0 = nTile * BN;

    v8f acc[4][2];
    #pragma unroll
    for (int i = 0; i < 4; ++i)
        #pragma unroll
        for (int j = 0; j < 2; ++j)
            acc[i][j] = (v8f){0.f, 0.f, 0.f, 0.f, 0.f, 0.f, 0.f, 0.f};

    const int KT = IN_FEATURES / BK;          // 64 K-tiles

#if USE_TDM
    if (wave == 0) {  // one wave drives the DMA engine (EXEC-independent op)
        tdm_load_tile_f16((unsigned)(uintptr_t)&sA[0][0],
                          xh + (size_t)m0 * IN_FEATURES,
                          IN_FEATURES, M_TOTAL, BK, BM, IN_FEATURES);
        tdm_load_tile_f16((unsigned)(uintptr_t)&sB[0][0],
                          wh + (size_t)n0 * IN_FEATURES,
                          IN_FEATURES, OUT_FEATURES, BK, BN, IN_FEATURES);
    }
#endif

    for (int kt = 0; kt < KT; ++kt) {
        const int cur = kt & 1;

#if USE_TDM
        if (wave == 0) {
            if (kt + 1 < KT) {   // prefetch next K-tile into the other buffer
                const int kb2 = (kt + 1) * BK;
                tdm_load_tile_f16((unsigned)(uintptr_t)&sA[cur ^ 1][0],
                                  xh + (size_t)m0 * IN_FEATURES + kb2,
                                  IN_FEATURES, M_TOTAL, BK, BM, IN_FEATURES);
                tdm_load_tile_f16((unsigned)(uintptr_t)&sB[cur ^ 1][0],
                                  wh + (size_t)n0 * IN_FEATURES + kb2,
                                  IN_FEATURES, OUT_FEATURES, BK, BN, IN_FEATURES);
                __builtin_amdgcn_s_wait_tensorcnt(2);  // current tile done (in-order)
            } else {
                __builtin_amdgcn_s_wait_tensorcnt(0);
            }
        }
        __syncthreads();                      // release compute waves
#else
        // Cooperative-copy fallback: 1024 16B chunks per tile, 4 per thread.
        __syncthreads();
        const int kb = kt * BK;
        #pragma unroll
        for (int c = 0; c < 4; ++c) {
            const int chunk = tid + 256 * c;
            const int row = chunk >> 3, c8 = chunk & 7;
            *(v8h*)(&sA[cur][row * LDS_STRIDE + c8 * 8]) =
                *(const v8h*)(xh + (size_t)(m0 + row) * IN_FEATURES + kb + c8 * 8);
            *(v8h*)(&sB[cur][row * LDS_STRIDE + c8 * 8]) =
                *(const v8h*)(wh + (size_t)(n0 + row) * IN_FEATURES + kb + c8 * 8);
        }
        __syncthreads();
#endif

        const _Float16* Ab = sA[cur];
        const _Float16* Bb = sB[cur];

        #pragma unroll
        for (int kk = 0; kk < BK; kk += 32) {
            const int mrow   = lane & 15;
            const int khalfA = (lane >> 4) * 8;    // A: upper half-wave K+8
            const int khalfB = (lane >> 4) * 8;    // B: upper half-wave K+8

            v16h afrag[4];
            #pragma unroll
            for (int i = 0; i < 4; ++i) {
                const _Float16* ap =
                    &Ab[(waveM * 64 + i * 16 + mrow) * LDS_STRIDE + kk + khalfA];
                v8h lo = *(const v8h*)(ap);         // K base .. base+7
                v8h hi = *(const v8h*)(ap + 16);    // K base+16 .. base+23
                afrag[i] = __builtin_shufflevector(lo, hi,
                    0, 1, 2, 3, 4, 5, 6, 7, 8, 9, 10, 11, 12, 13, 14, 15);
            }
            v16h bfrag[2];
            #pragma unroll
            for (int j = 0; j < 2; ++j) {
                const _Float16* bp =
                    &Bb[(waveN * 32 + j * 16 + mrow) * LDS_STRIDE + kk + khalfB];
                v8h lo = *(const v8h*)(bp);
                v8h hi = *(const v8h*)(bp + 16);
                bfrag[j] = __builtin_shufflevector(lo, hi,
                    0, 1, 2, 3, 4, 5, 6, 7, 8, 9, 10, 11, 12, 13, 14, 15);
            }
            #pragma unroll
            for (int i = 0; i < 4; ++i)
                #pragma unroll
                for (int j = 0; j < 2; ++j)
                    acc[i][j] = __builtin_amdgcn_wmma_f32_16x16x32_f16(
                        false, afrag[i], false, bfrag[j],
                        (short)0, acc[i][j], false, false);
        }
#if USE_TDM
        __syncthreads();   // all waves finished with buf `cur` before wave0
                           // issues the DMA that overwrites it next iteration
#endif
    }

    // Epilogue: C layout VGPR r -> M=r (+8 upper half-wave), lane -> N.
    const int nlocal = lane & 15;
    const int moff   = (lane >> 4) * 8;
    #pragma unroll
    for (int j = 0; j < 2; ++j) {
        const int n = n0 + waveN * 32 + j * 16 + nlocal;
        const float bv = bias[n];
        #pragma unroll
        for (int i = 0; i < 4; ++i) {
            const int mbase = m0 + waveM * 64 + i * 16 + moff;
            #pragma unroll
            for (int r = 0; r < 8; ++r)
                out[(size_t)(mbase + r) * OUT_FEATURES + n] = acc[i][j][r] + bv;
        }
    }
}

// ----------------------------------------------- fused fallback (small ws) --
__global__ __launch_bounds__(256)
void qlinear_fused_kernel(const float* __restrict__ x, const int* __restrict__ wq,
                          const float* __restrict__ absmax,
                          const float* __restrict__ bias, float* __restrict__ out)
{
    __shared__ _Float16 sA[BM * LDS_STRIDE];
    __shared__ _Float16 sB[BN * LDS_STRIDE];
    const int tid = threadIdx.x, lane = tid & 31, wave = tid >> 5;
    const int waveM = wave >> 2, waveN = wave & 3;
    const int mTiles = M_TOTAL / BM;
    const int nTile = blockIdx.x / mTiles, mTile = blockIdx.x % mTiles;
    const int m0 = mTile * BM, n0 = nTile * BN;
    v8f acc[4][2];
    #pragma unroll
    for (int i = 0; i < 4; ++i)
        #pragma unroll
        for (int j = 0; j < 2; ++j)
            acc[i][j] = (v8f){0.f, 0.f, 0.f, 0.f, 0.f, 0.f, 0.f, 0.f};
    const int c4 = tid & 15, r0 = tid >> 4;
    const float invRange = 1.0f / 7.0f;
    for (int kb = 0; kb < IN_FEATURES; kb += BK) {
        __syncthreads();
        #pragma unroll
        for (int i = 0; i < 8; ++i) {
            const int r = r0 + 16 * i;
            const float4 v = *(const float4*)(x + (size_t)(m0 + r) * IN_FEATURES + kb + c4 * 4);
            v4h ha; ha[0] = (_Float16)v.x; ha[1] = (_Float16)v.y;
            ha[2] = (_Float16)v.z; ha[3] = (_Float16)v.w;
            *(v4h*)(&sA[r * LDS_STRIDE + c4 * 4]) = ha;
            const int4 q = *(const int4*)(wq + (size_t)(n0 + r) * IN_FEATURES + kb + c4 * 4);
            const float s =
                absmax[(size_t)(n0 + r) * (IN_FEATURES / QBLOCK) + (kb >> 6)] * invRange;
            v4h hb; hb[0] = (_Float16)((float)q.x * s); hb[1] = (_Float16)((float)q.y * s);
            hb[2] = (_Float16)((float)q.z * s); hb[3] = (_Float16)((float)q.w * s);
            *(v4h*)(&sB[r * LDS_STRIDE + c4 * 4]) = hb;
        }
        __syncthreads();
        #pragma unroll
        for (int kk = 0; kk < BK; kk += 32) {
            const int mrow = lane & 15, khA = (lane >> 4) * 8, khB = (lane >> 4) * 8;
            v16h afrag[4];
            #pragma unroll
            for (int i = 0; i < 4; ++i) {
                const _Float16* ap = &sA[(waveM * 64 + i * 16 + mrow) * LDS_STRIDE + kk + khA];
                v8h lo = *(const v8h*)(ap), hi = *(const v8h*)(ap + 16);
                afrag[i] = __builtin_shufflevector(lo, hi,
                    0, 1, 2, 3, 4, 5, 6, 7, 8, 9, 10, 11, 12, 13, 14, 15);
            }
            v16h bfrag[2];
            #pragma unroll
            for (int j = 0; j < 2; ++j) {
                const _Float16* bp = &sB[(waveN * 32 + j * 16 + mrow) * LDS_STRIDE + kk + khB];
                v8h lo = *(const v8h*)(bp), hi = *(const v8h*)(bp + 16);
                bfrag[j] = __builtin_shufflevector(lo, hi,
                    0, 1, 2, 3, 4, 5, 6, 7, 8, 9, 10, 11, 12, 13, 14, 15);
            }
            #pragma unroll
            for (int i = 0; i < 4; ++i)
                #pragma unroll
                for (int j = 0; j < 2; ++j)
                    acc[i][j] = __builtin_amdgcn_wmma_f32_16x16x32_f16(
                        false, afrag[i], false, bfrag[j], (short)0, acc[i][j], false, false);
        }
    }
    const int nlocal = lane & 15, moff = (lane >> 4) * 8;
    #pragma unroll
    for (int j = 0; j < 2; ++j) {
        const int n = n0 + waveN * 32 + j * 16 + nlocal;
        const float bv = bias[n];
        #pragma unroll
        for (int i = 0; i < 4; ++i) {
            const int mbase = m0 + waveM * 64 + i * 16 + moff;
            #pragma unroll
            for (int r = 0; r < 8; ++r)
                out[(size_t)(mbase + r) * OUT_FEATURES + n] = acc[i][j][r] + bv;
        }
    }
}

// ---------------------------------------------------------------------------
extern "C" void kernel_launch(void* const* d_in, const int* in_sizes, int n_in,
                              void* d_out, int out_size, void* d_ws, size_t ws_size,
                              hipStream_t stream) {
    const float* x      = (const float*)d_in[0];
    const int*   wq     = (const int*)d_in[1];
    const float* absmax = (const float*)d_in[2];
    const float* bias   = (const float*)d_in[3];
    float*       out    = (float*)d_out;

    const int gemmGrid = (OUT_FEATURES / BN) * (M_TOTAL / BM);   // 86 * 64 = 5504
    const size_t xh_bytes = (size_t)M_TOTAL * IN_FEATURES * sizeof(_Float16);      // 64 MB
    const size_t wh_bytes = (size_t)OUT_FEATURES * IN_FEATURES * sizeof(_Float16); // 86 MB

    if (ws_size >= xh_bytes + wh_bytes) {
        _Float16* xh = (_Float16*)d_ws;
        _Float16* wh = (_Float16*)((char*)d_ws + xh_bytes);
        cvt_x_kernel<<<(M_TOTAL * IN_FEATURES) / (8 * 256), 256, 0, stream>>>(x, xh);
        dequant_w_kernel<<<(OUT_FEATURES * IN_FEATURES) / (16 * 256), 256, 0, stream>>>(
            wq, absmax, wh);
        qgemm_f16_wmma<<<gemmGrid, 256, 0, stream>>>(xh, wh, bias, out);
    } else {
        qlinear_fused_kernel<<<gemmGrid, 256, 0, stream>>>(x, wq, absmax, bias, out);
    }
}
